// NodeMamba_30227979829543
// MI455X (gfx1250) — compile-verified, matrix operand order checked by reading, not run on hardware
//
#include <hip/hip_runtime.h>
#include <hip/hip_bf16.h>
#include <math.h>

#define LTOK   8192
#define EDIM   512
#define DSTATE 16
#define DCONV  4
#define DTRANK 32
#define NNODE_ 2048

typedef __attribute__((ext_vector_type(16))) __bf16 v16bf;
typedef __attribute__((ext_vector_type(8)))  __bf16 v8bf;
typedef __attribute__((ext_vector_type(8)))  float  v8f;

// ---------------------------------------------------------------------------
// Activation epilogues: 0 = none, 1 = leaky-relu(0.01), 2 = softplus
// ---------------------------------------------------------------------------
__device__ __forceinline__ float act_apply(float v, int act) {
  if (act == 1) return v >= 0.f ? v : 0.01f * v;
  if (act == 2) return (v > 20.f) ? v : log1pf(expf(v));
  return v;
}

__device__ __forceinline__ v16bf frag_load(const __bf16* p) {
  // One wave32 16-bit fragment = two contiguous 16-byte chunks per lane
  // (CDNA5 ISA 7.12.2: VGPRs 0-3 hold K=khalf..khalf+7, VGPRs 4-7 hold +16).
  v8bf lo = *(const v8bf*)(p);
  v8bf hi = *(const v8bf*)(p + 16);
  return __builtin_shufflevector(lo, hi, 0, 1, 2, 3, 4, 5, 6, 7,
                                 8, 9, 10, 11, 12, 13, 14, 15);
}

// ---------------------------------------------------------------------------
// WMMA GEMM: C(M,N) = act( A(M,K; lda) * W(N,K)^T + bias )
// One wave per 16x64 output strip: 4 f32 accumulators share one A fragment
// -> 4 back-to-back v_wmma per 32-wide K step. K-loop is register
// double-buffered so next-step b128 loads are in flight while the current
// WMMAs execute (partial loadcnt waits instead of full drains).
// Waves within a block share the same N strip -> B (weight) fragments hit
// in the near cache while A streams.
// ---------------------------------------------------------------------------
__global__ void gemm_wmma_kernel(const __bf16* __restrict__ A, int lda,
                                 const __bf16* __restrict__ W,
                                 const float*  __restrict__ bias,
                                 float*  __restrict__ Cf,
                                 __bf16* __restrict__ Cb, int ldc,
                                 int M, int N, int K, int act)
{
  const int lane   = threadIdx.x & 31;
  const int wave   = (blockIdx.x * blockDim.x + threadIdx.x) >> 5;
  const int tilesM = M >> 4;
  const int tilesN = N >> 6;                       // 64-wide wave strips
  if (wave >= tilesM * tilesN) return;             // uniform per wave
  const int tm    = wave % tilesM;                 // M fast-varying:
  const int tn    = wave / tilesM;                 // block shares B strip
  const int l16   = lane & 15;
  const int khalf = (lane >> 4) << 3;              // 0 or 8

  const __bf16* arow  = A + (long)(tm * 16 + l16) * lda + khalf;
  const __bf16* wrow  = W + (long)(tn * 64 + l16) * K + khalf;
  const long    wstep = (long)16 * K;              // next 16-column group

  v8f acc0 = {}, acc1 = {}, acc2 = {}, acc3 = {};
  v16bf a  = frag_load(arow);
  v16bf b0 = frag_load(wrow);
  v16bf b1 = frag_load(wrow + wstep);
  v16bf b2 = frag_load(wrow + 2 * wstep);
  v16bf b3 = frag_load(wrow + 3 * wstep);

  const int steps = K >> 5;
  for (int it = 0; it < steps - 1; ++it) {
    arow += 32;
    wrow += 32;
    // issue next-step loads before consuming current fragments
    const v16bf an  = frag_load(arow);
    const v16bf bn0 = frag_load(wrow);
    const v16bf bn1 = frag_load(wrow + wstep);
    const v16bf bn2 = frag_load(wrow + 2 * wstep);
    const v16bf bn3 = frag_load(wrow + 3 * wstep);
    acc0 = __builtin_amdgcn_wmma_f32_16x16x32_bf16(false, a, false, b0, (short)0, acc0, false, false);
    acc1 = __builtin_amdgcn_wmma_f32_16x16x32_bf16(false, a, false, b1, (short)0, acc1, false, false);
    acc2 = __builtin_amdgcn_wmma_f32_16x16x32_bf16(false, a, false, b2, (short)0, acc2, false, false);
    acc3 = __builtin_amdgcn_wmma_f32_16x16x32_bf16(false, a, false, b3, (short)0, acc3, false, false);
    a = an; b0 = bn0; b1 = bn1; b2 = bn2; b3 = bn3;
  }
  acc0 = __builtin_amdgcn_wmma_f32_16x16x32_bf16(false, a, false, b0, (short)0, acc0, false, false);
  acc1 = __builtin_amdgcn_wmma_f32_16x16x32_bf16(false, a, false, b1, (short)0, acc1, false, false);
  acc2 = __builtin_amdgcn_wmma_f32_16x16x32_bf16(false, a, false, b2, (short)0, acc2, false, false);
  acc3 = __builtin_amdgcn_wmma_f32_16x16x32_bf16(false, a, false, b3, (short)0, acc3, false, false);

  v8f accs[4] = {acc0, acc1, acc2, acc3};
#pragma unroll
  for (int j = 0; j < 4; ++j) {
    const int   col = tn * 64 + j * 16 + l16;
    const float bv  = bias ? bias[col] : 0.f;
#pragma unroll
    for (int r = 0; r < 8; ++r) {
      const int row = tm * 16 + r + ((lane >> 4) << 3);   // C/D layout 7.12.2
      const float v = act_apply(accs[j][r] + bv, act);
      if (Cf) Cf[(long)row * ldc + col] = v;
      if (Cb) Cb[(long)row * ldc + col] = (__bf16)v;
    }
  }
}

// ---------------------------------------------------------------------------
// One-shot f32 -> bf16 staging (weights + concat activation matrix).
// ---------------------------------------------------------------------------
__global__ void cvt_bf16_kernel(const float* __restrict__ in,
                                __bf16* __restrict__ out, long n)
{
  const long i = (long)blockIdx.x * blockDim.x + threadIdx.x;
  if (i < n) out[i] = (__bf16)in[i];
}

__global__ void concat_bf16_kernel(const float* __restrict__ q,
                                   const float* __restrict__ kv,
                                   const float* __restrict__ ke,
                                   __bf16* __restrict__ A)
{
  const long i = (long)blockIdx.x * blockDim.x + threadIdx.x;
  if (i >= (long)LTOK * 3 * EDIM) return;
  const int t   = (int)(i / (3 * EDIM));
  const int col = (int)(i % (3 * EDIM));
  float v;
  if (col < EDIM)            v = q [(long)t * EDIM + col];
  else if (col < 2 * EDIM)   v = kv[(long)t * EDIM + (col - EDIM)];
  else                       v = ke[(long)t * EDIM + (col - 2 * EDIM)];
  A[i] = (__bf16)v;
}

// ---------------------------------------------------------------------------
// Segment-masked depthwise causal conv (DCONV=4) + SiLU.
// Writes f32 (scan input) and bf16 (dbc GEMM input).
// ---------------------------------------------------------------------------
__global__ void conv_silu_kernel(const float* __restrict__ xz,
                                 const int*   __restrict__ eb,
                                 const float* __restrict__ conv_w,
                                 const float* __restrict__ conv_b,
                                 float*  __restrict__ xc_f,
                                 __bf16* __restrict__ xc_b)
{
  const long idx = (long)blockIdx.x * blockDim.x + threadIdx.x;
  if (idx >= (long)LTOK * EDIM) return;
  const int t = (int)(idx / EDIM);
  const int c = (int)(idx % EDIM);
  const int b = eb[t];
  float acc = conv_b[c];
#pragma unroll
  for (int k = 0; k < DCONV; ++k) {
    const int tt = t - (DCONV - 1) + k;
    if (tt >= 0 && eb[tt] == b)
      acc += conv_w[c * DCONV + k] * xz[(long)tt * (2 * EDIM) + c];
  }
  acc = acc / (1.f + expf(-acc));   // silu
  xc_f[idx] = acc;
  xc_b[idx] = (__bf16)acc;
}

// ---------------------------------------------------------------------------
// Selective scan. Block = 256 lanes = 16 channels x 16 states; 32 blocks.
// 16-state reduction via half-wave shfl_xor butterflies; fuses
// y = (scan + D*xc) * silu(z), written directly as bf16 for the out-proj GEMM.
// ---------------------------------------------------------------------------
__global__ void scan_kernel(const float* __restrict__ delta,  // L x 512
                            const float* __restrict__ xc,     // L x 512
                            const float* __restrict__ dbc,    // L x 64 (B@32, C@48)
                            const float* __restrict__ xz,     // L x 1024 (z @ +512)
                            const int*   __restrict__ eb,
                            const float* __restrict__ A_log,  // 512 x 16
                            const float* __restrict__ Dvec,   // 512
                            __bf16* __restrict__ y)           // L x 512 (bf16)
{
  const int tid = threadIdx.x;
  const int s   = tid & 15;
  const int c   = blockIdx.x * 16 + (tid >> 4);
  const float Acs = -expf(A_log[c * DSTATE + s]);
  const float Dc  = Dvec[c];
  float h = 0.f;
  int prev_b = -1;
  for (int t = 0; t < LTOK; ++t) {
    const int   b   = eb[t];
    const float dlt = delta[(long)t * EDIM + c];
    const float xcv = xc[(long)t * EDIM + c];
    const float Bv  = dbc[(long)t * 64 + DTRANK + s];
    const float Cv  = dbc[(long)t * 64 + DTRANK + DSTATE + s];
    const bool first = (b != prev_b);
    prev_b = b;
    const float dA = expf(dlt * Acs);
    h = (first ? 0.f : dA * h) + dlt * Bv * xcv;
    float part = h * Cv;
    part += __shfl_xor(part, 1, 32);
    part += __shfl_xor(part, 2, 32);
    part += __shfl_xor(part, 4, 32);
    part += __shfl_xor(part, 8, 32);
    if (s == 0) {
      const float zv = xz[(long)t * (2 * EDIM) + EDIM + c];
      float yv = part + Dc * xcv;
      yv *= zv / (1.f + expf(-zv));
      y[(long)t * EDIM + c] = (__bf16)yv;
    }
  }
}

// ---------------------------------------------------------------------------
// Zero + atomic segment-sum into nodes.
// ---------------------------------------------------------------------------
__global__ void zero_kernel(float* __restrict__ p, long n) {
  const long i = (long)blockIdx.x * blockDim.x + threadIdx.x;
  if (i < n) p[i] = 0.f;
}

__global__ void segsum_kernel(const float* __restrict__ tok,
                              const int*   __restrict__ index,
                              float* __restrict__ out)
{
  const long i = (long)blockIdx.x * blockDim.x + threadIdx.x;
  if (i >= (long)LTOK * EDIM) return;
  const int t = (int)(i / EDIM);
  const int c = (int)(i % EDIM);
  atomicAdd(&out[(long)index[t] * EDIM + c], tok[i]);
}

// ---------------------------------------------------------------------------
// Host launcher
// ---------------------------------------------------------------------------
extern "C" void kernel_launch(void* const* d_in, const int* in_sizes, int n_in,
                              void* d_out, int out_size, void* d_ws, size_t ws_size,
                              hipStream_t stream)
{
  const float* q          = (const float*)d_in[0];
  const float* k_v        = (const float*)d_in[1];
  const float* k_e        = (const float*)d_in[2];
  const int*   index      = (const int*)  d_in[3];
  /* d_in[4] = nnode scalar (2048) */
  const int*   edge_batch = (const int*)  d_in[5];
  const float* w_weight   = (const float*)d_in[6];
  const float* w_bias     = (const float*)d_in[7];
  const float* in_proj_w  = (const float*)d_in[8];
  const float* conv_w     = (const float*)d_in[9];
  const float* conv_b     = (const float*)d_in[10];
  const float* x_proj_w   = (const float*)d_in[11];
  const float* dt_proj_w  = (const float*)d_in[12];
  const float* dt_proj_b  = (const float*)d_in[13];
  const float* A_log      = (const float*)d_in[14];
  const float* Dvec       = (const float*)d_in[15];
  const float* out_proj_w = (const float*)d_in[16];
  float* out = (float*)d_out;

  // --- workspace carve-up (256B aligned), with lifetime-based overlays ---
  char*  base = (char*)d_ws;
  size_t off  = 0;
  auto alloc = [&](size_t bytes) {
    char* p = base + off;
    off += (bytes + 255) & ~(size_t)255;
    return p;
  };
  float*  xz_f  = (float*) alloc((size_t)LTOK * 2 * EDIM * 4); // 32MB region R0
  __bf16* Abf   = (__bf16*)xz_f;            // concat input; dead before xz write
  __bf16* x_bf  = (__bf16*)alloc((size_t)LTOK * EDIM * 2);     // 8MB  region R1
  __bf16* y_bf  = x_bf;                     // x dead after xz GEMM
  __bf16* wwb   = (__bf16*)alloc((size_t)EDIM * 3 * EDIM * 2);
  __bf16* ipb   = (__bf16*)alloc((size_t)2 * EDIM * EDIM * 2);
  __bf16* xpb   = (__bf16*)alloc((size_t)64 * EDIM * 2);
  __bf16* dtb   = (__bf16*)alloc((size_t)EDIM * DTRANK * 2);
  __bf16* opb   = (__bf16*)alloc((size_t)EDIM * EDIM * 2);
  float*  xc_f  = (float*) alloc((size_t)LTOK * EDIM * 4);
  __bf16* xc_b  = (__bf16*)alloc((size_t)LTOK * EDIM * 2);
  float*  dbc_f = (float*) alloc((size_t)LTOK * 64 * 4);
  __bf16* dbc_b = (__bf16*)alloc((size_t)LTOK * 64 * 2);
  float*  dlt_f = (float*) alloc((size_t)LTOK * EDIM * 4);
  float*  tok_f = dlt_f;                    // delta dead after scan

  auto cvt = [&](const float* in, __bf16* outp, long n) {
    hipLaunchKernelGGL(cvt_bf16_kernel, dim3((unsigned)((n + 255) / 256)),
                       dim3(256), 0, stream, in, outp, n);
  };
  auto gemm = [&](const __bf16* A, int lda, const __bf16* W, const float* bias,
                  float* Cf, __bf16* Cb, int ldc, int M, int N, int K, int act) {
    const int waves  = (M / 16) * (N / 64);
    const int blocks = (waves * 32 + 255) / 256;
    hipLaunchKernelGGL(gemm_wmma_kernel, dim3(blocks), dim3(256), 0, stream,
                       A, lda, W, bias, Cf, Cb, ldc, M, N, K, act);
  };

  // 0) stage weights + concat activations as bf16 (one-time per launch)
  cvt(w_weight,   wwb, (long)EDIM * 3 * EDIM);
  cvt(in_proj_w,  ipb, (long)2 * EDIM * EDIM);
  cvt(x_proj_w,   xpb, (long)64 * EDIM);
  cvt(dt_proj_w,  dtb, (long)EDIM * DTRANK);
  cvt(out_proj_w, opb, (long)EDIM * EDIM);
  hipLaunchKernelGGL(concat_bf16_kernel,
                     dim3((unsigned)(((long)LTOK * 3 * EDIM + 255) / 256)),
                     dim3(256), 0, stream, q, k_v, k_e, Abf);

  // 1) x = leaky_relu(concat @ Ww^T + b)           M=8192 N=512  K=1536 -> bf16
  gemm(Abf, 3 * EDIM, wwb, w_bias, nullptr, x_bf, EDIM, LTOK, EDIM, 3 * EDIM, 1);
  // 2) xz = x @ in_proj_w^T                        M=8192 N=1024 K=512  -> f32
  gemm(x_bf, EDIM, ipb, nullptr, xz_f, nullptr, 2 * EDIM, LTOK, 2 * EDIM, EDIM, 0);
  // 3) xc = silu(segment_conv(xz[:, :512]))        -> f32 + bf16
  hipLaunchKernelGGL(conv_silu_kernel, dim3((LTOK * EDIM) / 256), dim3(256), 0,
                     stream, xz_f, edge_batch, conv_w, conv_b, xc_f, xc_b);
  // 4) dbc = xc @ x_proj_w^T                       M=8192 N=64   K=512  -> f32 + bf16
  gemm(xc_b, EDIM, xpb, nullptr, dbc_f, dbc_b, 64, LTOK, 64, EDIM, 0);
  // 5) delta = softplus(dbc[:, :32] @ dt^T + b)    M=8192 N=512  K=32   -> f32
  gemm(dbc_b, 64, dtb, dt_proj_b, dlt_f, nullptr, EDIM, LTOK, EDIM, DTRANK, 2);
  // 6) selective scan (fuses +D*xc and *silu(z))   -> y bf16
  hipLaunchKernelGGL(scan_kernel, dim3(EDIM / 16), dim3(256), 0, stream,
                     dlt_f, xc_f, dbc_f, xz_f, edge_batch, A_log, Dvec, y_bf);
  // 7) tok = y @ out_proj_w^T                      M=8192 N=512  K=512  -> f32
  gemm(y_bf, EDIM, opb, nullptr, tok_f, nullptr, EDIM, LTOK, EDIM, EDIM, 0);
  // 8) node segment-sum
  hipLaunchKernelGGL(zero_kernel, dim3((NNODE_ * EDIM) / 256), dim3(256), 0,
                     stream, out, (long)NNODE_ * EDIM);
  hipLaunchKernelGGL(segsum_kernel, dim3((LTOK * EDIM) / 256), dim3(256), 0,
                     stream, tok_f, index, out);
}
